// WordCharLSTMCRF_57604101374149
// MI455X (gfx1250) — matrix operand order repaired and checked by reading, not compiled, not run on hardware
//
#include <hip/hip_runtime.h>
#include <hip/hip_bf16.h>

// WordCharLSTMCRF forward for MI455X (gfx1250, wave32, WMMA).
// Pipeline:
//   1) char vocab gate table  CT[128][256] = char_emb @ Wih_c.T + b_c   (vocab tiny -> table)
//   2) pack weights to f16 (word W_hh in B-fragment-tiled layout); pack padded
//      word-lstm input X[8192][384] (300 emb | pad | 64 charfeat | pad)
//   3) char LSTM recurrence via v_wmma_f32_16x16x32_f16 (h in LDS f16, c in regs),
//      writes char feats into X
//   4) word gate GEMM WG = X @ WihPad.T + b_w  (LDS-staged WMMA GEMM, K=384),
//      output stored in D-fragment-tiled layout (tile base + e*32 + lane)
//   5) word LSTM recurrence (1 workgroup, 32 waves, fully unrolled WMMA K-loop,
//      base+imm-offset gate/weight loads, prefetch of next step) + inline logits
//   6) CRF log-likelihood (scalar), out[0] = -sum(ll)

typedef _Float16 f16_t;
typedef __attribute__((ext_vector_type(16))) _Float16 v16h;
typedef __attribute__((ext_vector_type(8)))  float    v8f;

#define T_SEQ   128
#define B_BATCH 64
#define W_CHARS 16
#define EW      300
#define EC      50
#define HC      64
#define HW      256
#define K_TAGS  24
#define KPAD    384      // padded word-lstm input width
#define NWORDS  8192     // B*T

__device__ __forceinline__ v8f wmma_f16(v16h a, v16h b, v8f c) {
  // v_wmma_f32_16x16x32_f16: D = A(16x32) x B(32x16) + C
  return __builtin_amdgcn_wmma_f32_16x16x32_f16(false, a, false, b, (short)0, c,
                                                false, false);
}

// A fragment (16x32 f16), source row-major MxK, p -> element (m0, k0), ld in elems.
// Lane l: M = l%16; lanes 0-15 hold K=0..15, lanes 16-31 hold K=16..31 (contiguous).
__device__ __forceinline__ v16h load_frag_a(const f16_t* p, int ld) {
  const int lane = threadIdx.x & 31;
  const int hs = (lane >> 4) & 1;
  const f16_t* r = p + (lane & 15) * ld + 16 * hs;
  v16h a;
#pragma unroll
  for (int i = 0; i < 8; ++i) a[i] = r[i];
#pragma unroll
  for (int i = 0; i < 8; ++i) a[8 + i] = r[8 + i];
  return a;
}

// B fragment (32x16 f16) from an N-major weight matrix w[n][k]; p -> (n0, k0).
// Lane l: N = l%16; lanes 0-15 hold K=0..15, lanes 16-31 hold K=16..31 (contiguous).
__device__ __forceinline__ v16h load_frag_b(const f16_t* p, int ld) {
  const int lane = threadIdx.x & 31;
  const int hs = (lane >> 4) & 1;
  const f16_t* r = p + (lane & 15) * ld + 16 * hs;
  v16h b;
#pragma unroll
  for (int i = 0; i < 16; ++i) b[i] = r[i];
  return b;
}

// B fragment from fragment-tiled storage: tile base p, lane's 16 halves contiguous.
__device__ __forceinline__ v16h load_frag_b_tiled(const f16_t* p) {
  const f16_t* r = p + (threadIdx.x & 31) * 16;
  v16h b;
#pragma unroll
  for (int i = 0; i < 16; ++i) b[i] = r[i];
  return b;
}

__device__ __forceinline__ float sigf(float x) { return 1.0f / (1.0f + expf(-x)); }

// ---------------- prep kernels ----------------

__global__ void k_char_table(const float* __restrict__ char_emb,
                             const float* __restrict__ w_ih,
                             const float* __restrict__ bias,
                             float* __restrict__ CT) {
  const int row = blockIdx.x;    // char vocab 0..127
  const int col = threadIdx.x;   // gate 0..255
  float s = bias[col];
  for (int e = 0; e < EC; ++e) s += char_emb[row * EC + e] * w_ih[col * EC + e];
  CT[row * 256 + col] = s;
}

__global__ void k_cvt_f16(const float* __restrict__ src, f16_t* __restrict__ dst,
                          int n) {
  int i = blockIdx.x * blockDim.x + threadIdx.x;
  if (i < n) dst[i] = (f16_t)src[i];
}

// word W_hh (1024x256) -> B-fragment-tiled f16:
// tile (N=n/16, Kt=k/32): dst[((N*8+Kt)*32 + lane)*16 + ki], lane = (k%32)/16*16 + n%16
__global__ void k_pack_whh_frag(const float* __restrict__ w,
                                f16_t* __restrict__ dst) {
  int i = blockIdx.x * blockDim.x + threadIdx.x;
  if (i >= 1024 * 256) return;
  const int n = i >> 8, k = i & 255;
  const int N = n >> 4, ln = n & 15;
  const int kt = k >> 5, ko = k & 31;
  const int lane = ((ko >> 4) << 4) | ln;
  const int ki = ko & 15;
  dst[(((N * 8 + kt) * 32 + lane) << 4) + ki] = (f16_t)w[i];
}

__global__ void k_pack_wih(const float* __restrict__ w, f16_t* __restrict__ dst) {
  int i = blockIdx.x * blockDim.x + threadIdx.x;
  if (i >= 1024 * KPAD) return;
  const int r = i / KPAD, c = i % KPAD;
  float v = 0.0f;
  if (c < 300) v = w[r * 364 + c];                                 // word-emb cols
  else if (c >= 304 && c < 368) v = w[r * 364 + 300 + (c - 304)];  // char cols
  dst[i] = (f16_t)v;
}

__global__ void k_pack_x(const int* __restrict__ word_inputs,
                         const float* __restrict__ word_emb,
                         f16_t* __restrict__ X) {
  int i = blockIdx.x * blockDim.x + threadIdx.x;
  if (i >= NWORDS * KPAD) return;
  const int r = i / KPAD, c = i % KPAD;   // r = t*B + b
  float v = 0.0f;
  if (c < 300) v = word_emb[word_inputs[r] * EW + c];
  X[i] = (f16_t)v;   // char-feat cols (304..367) overwritten by k_char_lstm
}

// ---------------- char LSTM (128 blocks x 64 words, WMMA recurrence) ----------------

__global__ __launch_bounds__(256) void k_char_lstm(
    const int* __restrict__ char_inputs, const float* __restrict__ CT,
    const f16_t* __restrict__ whh, f16_t* __restrict__ Xpad) {
  __shared__ __attribute__((aligned(16))) f16_t s_whh[256 * 64];
  __shared__ __attribute__((aligned(16))) f16_t s_h[64 * 72];
  __shared__ int s_cid[64];
  const int tid = threadIdx.x;
  const int wave = tid >> 5, lane = tid & 31;
  const int hs = (lane >> 4) & 1, lm = lane & 15;
  const int mt = wave & 3;       // row tile (16 words)
  const int j0 = wave >> 2;      // handles j = j0 and j0+2, j in [0,4)
  const int word0 = blockIdx.x * 64;

  for (int i = tid; i < 256 * 64; i += 256) s_whh[i] = whh[i];
  for (int i = tid; i < 64 * 72; i += 256) s_h[i] = (f16_t)0.0f;
  __syncthreads();

  float creg[2][8];
#pragma unroll
  for (int q = 0; q < 2; ++q)
#pragma unroll
    for (int e = 0; e < 8; ++e) creg[q][e] = 0.0f;

  for (int t = 0; t < W_CHARS; ++t) {
    if (tid < 64) s_cid[tid] = char_inputs[(word0 + tid) * W_CHARS + t];
    __syncthreads();

    v8f acc[2][4];
#pragma unroll
    for (int q = 0; q < 2; ++q) {
      const int j = j0 + 2 * q;
      // C init = gathered input-gate table (includes bias)
#pragma unroll
      for (int g = 0; g < 4; ++g)
#pragma unroll
        for (int e = 0; e < 8; ++e) {
          const int m = mt * 16 + e + 8 * hs;
          acc[q][g][e] = CT[s_cid[m] * 256 + g * 64 + j * 16 + lm];
        }
#pragma unroll
      for (int kk = 0; kk < HC; kk += 32) {
        v16h a = load_frag_a(&s_h[(mt * 16) * 72 + kk], 72);
#pragma unroll
        for (int g = 0; g < 4; ++g) {
          v16h b = load_frag_b(&s_whh[(g * 64 + j * 16) * 64 + kk], 64);
          acc[q][g] = wmma_f16(a, b, acc[q][g]);
        }
      }
    }
    __syncthreads();   // all h reads done before rewrite
#pragma unroll
    for (int q = 0; q < 2; ++q) {
      const int j = j0 + 2 * q;
#pragma unroll
      for (int e = 0; e < 8; ++e) {
        const int m = mt * 16 + e + 8 * hs;
        const float iv = sigf(acc[q][0][e]);
        const float fv = sigf(acc[q][1][e]);
        const float gv = tanhf(acc[q][2][e]);
        const float ov = sigf(acc[q][3][e]);
        const float c = fv * creg[q][e] + iv * gv;
        creg[q][e] = c;
        s_h[m * 72 + j * 16 + lm] = (f16_t)(ov * tanhf(c));
      }
    }
    __syncthreads();
  }

  // final h -> char-feat columns of padded X; word n=b*T+t maps to x-row t*B+b
  for (int i = tid; i < 64 * 64; i += 256) {
    const int m = i >> 6, hcol = i & 63;
    const int n = word0 + m;
    const int b = n >> 7, tp = n & 127;
    Xpad[(tp * 64 + b) * KPAD + 304 + hcol] = s_h[m * 72 + hcol];
  }
}

// ---------------- word input-gate GEMM: WG = X @ WihPad.T + b ----------------
// Output stored fragment-tiled: tile (R=row/16, N=col/16) at ((R*64+N)*256 + e*32 + lane)

__global__ __launch_bounds__(256) void k_word_gates(
    const f16_t* __restrict__ X, const f16_t* __restrict__ Wp,
    const float* __restrict__ bias, float* __restrict__ WG) {
  __shared__ __attribute__((aligned(16))) f16_t sA[64 * 40];   // 64 x 32 (+pad)
  __shared__ __attribute__((aligned(16))) f16_t sB[256 * 32];  // 256 x 32
  const int tid = threadIdx.x;
  const int wave = tid >> 5, lane = tid & 31;
  const int lm = lane & 15;
  const int mt = wave & 3;
  const int ng = (wave >> 2) * 8;          // 8 n-tiles per wave
  const int row0 = (blockIdx.x >> 2) * 64;
  const int n0 = (blockIdx.x & 3) * 256;

  v8f acc[8];
#pragma unroll
  for (int s = 0; s < 8; ++s)
#pragma unroll
    for (int e = 0; e < 8; ++e) acc[s][e] = 0.0f;

  for (int kt = 0; kt < KPAD / 32; ++kt) {
    const int k0 = kt * 32;
    {  // stage A tile: 8 halves (16B) per thread
      const int r = tid >> 2, cs = (tid & 3) * 8;
      *(float4*)&sA[r * 40 + cs] = *(const float4*)&X[(row0 + r) * KPAD + k0 + cs];
      if (kt + 1 < KPAD / 32)
        __builtin_prefetch(&X[(row0 + r) * KPAD + k0 + 32 + cs], 0, 0);
    }
    {  // stage B tile: one 32-half weight row per thread
      const f16_t* src = &Wp[(n0 + tid) * KPAD + k0];
      f16_t* dst = &sB[tid * 32];
#pragma unroll
      for (int u = 0; u < 4; ++u)
        *(float4*)&dst[u * 8] = *(const float4*)&src[u * 8];
      if (kt + 1 < KPAD / 32) __builtin_prefetch(src + 32, 0, 0);
    }
    __syncthreads();
    v16h a = load_frag_a(&sA[(mt * 16) * 40], 40);
#pragma unroll
    for (int s = 0; s < 8; ++s) {
      v16h b = load_frag_b(&sB[((ng + s) * 16) * 32], 32);
      acc[s] = wmma_f16(a, b, acc[s]);
    }
    __syncthreads();
  }

  // fragment-tiled store: single base + immediate offsets
  float* dst =
      WG + (size_t)(((row0 >> 4) + mt) * 64 + (n0 >> 4) + ng) * 256 + lane;
#pragma unroll
  for (int s = 0; s < 8; ++s) {
    const float bv = bias[n0 + (ng + s) * 16 + lm];
#pragma unroll
    for (int e = 0; e < 8; ++e)
      dst[s * 256 + e * 32] = acc[s][e] + bv;
  }
}

// ---------------- word LSTM recurrence (+ inline logits projection) ----------------

__global__ __launch_bounds__(1024) void k_word_lstm(
    const float* __restrict__ WG, const f16_t* __restrict__ whhT,
    const float* __restrict__ proj_w, const float* __restrict__ proj_b,
    float* __restrict__ logits) {
  __shared__ __attribute__((aligned(16))) f16_t s_h[64 * 264];  // 64 x 256 (+pad)
  const int tid = threadIdx.x;
  const int wave = tid >> 5, lane = tid & 31;
  const int mt = wave & 3;           // batch-row tile
  const int j0 = (wave >> 2) * 2;    // j = j0, j0+1; j in [0,16)

  for (int i = tid; i < 64 * 264; i += 1024) s_h[i] = (f16_t)0.0f;
  __syncthreads();

  float creg[2][8];
#pragma unroll
  for (int q = 0; q < 2; ++q)
#pragma unroll
    for (int e = 0; e < 8; ++e) creg[q][e] = 0.0f;

  // per-wave constant bases into fragment-tiled WG / whhT
  const f16_t* wb_base = whhT + (size_t)j0 * 4096;  // + q*4096 + g*65536 + kt*512

  for (int t = 0; t < T_SEQ; ++t) {
    const float* wg_base = WG + (size_t)((t * 4 + mt) * 64) * 256 + lane;
    if (t + 1 < T_SEQ)
      __builtin_prefetch(WG + (size_t)(((t + 1) * 4 + mt) * 64 + j0) * 256 + lane,
                         0, 0);
    v8f acc[2][4];
#pragma unroll
    for (int q = 0; q < 2; ++q) {
      const int j = j0 + q;
      const float* wq = wg_base + j * 256;
#pragma unroll
      for (int g = 0; g < 4; ++g)
#pragma unroll
        for (int e = 0; e < 8; ++e)
          acc[q][g][e] = wq[g * 4096 + e * 32];   // tile N = g*16+j, imm offsets
#pragma unroll
      for (int kt = 0; kt < 8; ++kt) {
        v16h a = load_frag_a(&s_h[(mt * 16) * 264 + kt * 32], 264);
#pragma unroll
        for (int g = 0; g < 4; ++g) {
          v16h b = load_frag_b_tiled(wb_base + (size_t)q * 4096 +
                                     (size_t)g * 65536 + (size_t)kt * 512);
          acc[q][g] = wmma_f16(a, b, acc[q][g]);
        }
      }
    }
    __syncthreads();   // all h reads complete
#pragma unroll
    for (int q = 0; q < 2; ++q) {
      const int j = j0 + q;
#pragma unroll
      for (int e = 0; e < 8; ++e) {
        const int m = mt * 16 + e + 8 * ((lane >> 4) & 1);
        const float iv = sigf(acc[q][0][e]);
        const float fv = sigf(acc[q][1][e]);
        const float gv = tanhf(acc[q][2][e]);
        const float ov = sigf(acc[q][3][e]);
        const float c = fv * creg[q][e] + iv * gv;
        creg[q][e] = c;
        s_h[m * 264 + j * 16 + (lane & 15)] = (f16_t)(ov * tanhf(c));
      }
    }
    __syncthreads();
    // logits[t] = h @ proj_w.T + proj_b  (64x24, small -> VALU)
    for (int o = tid; o < 64 * K_TAGS; o += 1024) {
      const int b = o / K_TAGS, k = o % K_TAGS;
      float s = proj_b[k];
#pragma unroll 8
      for (int kk = 0; kk < HW; ++kk)
        s += (float)s_h[b * 264 + kk] * proj_w[k * HW + kk];
      logits[t * (64 * K_TAGS) + o] = s;
    }
    __syncthreads();
  }
}

// ---------------- CRF log-likelihood, out[0] = -sum_b ll(b) ----------------

__global__ __launch_bounds__(64) void k_crf(
    const float* __restrict__ logits, const int* __restrict__ tags,
    const float* __restrict__ trans, const float* __restrict__ start_t,
    const float* __restrict__ end_t, float* __restrict__ out) {
  const int b = threadIdx.x;
  float alpha[K_TAGS];
#pragma unroll
  for (int k = 0; k < K_TAGS; ++k) alpha[k] = start_t[k] + logits[b * K_TAGS + k];
  for (int t = 1; t < T_SEQ; ++t) {
    const int tg = tags[t * 64 + b];
    if (tg != 1) {  // mask: PAD == 1
      float na[K_TAGS];
      for (int j = 0; j < K_TAGS; ++j) {
        float mx = -3.0e38f;
#pragma unroll
        for (int i = 0; i < K_TAGS; ++i)
          mx = fmaxf(mx, alpha[i] + trans[i * K_TAGS + j]);
        float s = 0.0f;
#pragma unroll
        for (int i = 0; i < K_TAGS; ++i)
          s += expf(alpha[i] + trans[i * K_TAGS + j] - mx);
        na[j] = mx + logf(s) + logits[(t * 64 + b) * K_TAGS + j];
      }
#pragma unroll
      for (int k = 0; k < K_TAGS; ++k) alpha[k] = na[k];
    }
  }
  float mx = -3.0e38f;
#pragma unroll
  for (int k = 0; k < K_TAGS; ++k) mx = fmaxf(mx, alpha[k] + end_t[k]);
  float se = 0.0f;
#pragma unroll
  for (int k = 0; k < K_TAGS; ++k) se += expf(alpha[k] + end_t[k] - mx);
  const float denom = mx + logf(se);

  float emit = 0.0f, tsum = 0.0f;
  int prev = tags[b];
  int last_tag = prev;
  for (int t = 0; t < T_SEQ; ++t) {
    const int tg = tags[t * 64 + b];
    if (tg != 1) {  // padding is a contiguous suffix
      emit += logits[(t * 64 + b) * K_TAGS + tg];
      if (t > 0) tsum += trans[prev * K_TAGS + tg];
      last_tag = tg;
    }
    prev = tg;
  }
  const float ll = (start_t[tags[b]] + emit + tsum + end_t[last_tag]) - denom;

  __shared__ float red[64];
  red[b] = ll;
  __syncthreads();
  if (b == 0) {
    float a = 0.0f;
    for (int i = 0; i < 64; ++i) a += red[i];
    out[0] = -a;
  }
}

// ---------------- launch ----------------

extern "C" void kernel_launch(void* const* d_in, const int* in_sizes, int n_in,
                              void* d_out, int out_size, void* d_ws, size_t ws_size,
                              hipStream_t stream) {
  (void)in_sizes; (void)n_in; (void)out_size; (void)ws_size;
  const int*   word_inputs = (const int*)d_in[0];
  const int*   char_inputs = (const int*)d_in[1];
  const int*   tag_inputs  = (const int*)d_in[2];
  const float* word_emb    = (const float*)d_in[3];
  const float* char_emb    = (const float*)d_in[4];
  const float* char_w_ih   = (const float*)d_in[5];
  const float* char_w_hh   = (const float*)d_in[6];
  const float* char_b      = (const float*)d_in[7];
  const float* word_w_ih   = (const float*)d_in[8];
  const float* word_w_hh   = (const float*)d_in[9];
  const float* word_b      = (const float*)d_in[10];
  const float* proj_w      = (const float*)d_in[11];
  const float* proj_b      = (const float*)d_in[12];
  const float* trans       = (const float*)d_in[13];
  const float* start_t     = (const float*)d_in[14];
  const float* end_t       = (const float*)d_in[15];

  char* base = (char*)d_ws;
  size_t off = 0;
  auto bump = [&](size_t bytes) -> void* {
    void* p = base + off;
    off += (bytes + 255) & ~(size_t)255;
    return p;
  };
  float* CT     = (float*)bump(128 * 256 * sizeof(float));
  f16_t* whhC16 = (f16_t*)bump(256 * 64 * sizeof(f16_t));
  f16_t* whhT16 = (f16_t*)bump(1024 * 256 * sizeof(f16_t));
  f16_t* wihP16 = (f16_t*)bump((size_t)1024 * KPAD * sizeof(f16_t));
  f16_t* Xpad   = (f16_t*)bump((size_t)NWORDS * KPAD * sizeof(f16_t));
  float* WG     = (float*)bump((size_t)NWORDS * 1024 * sizeof(float));
  float* logits = (float*)bump((size_t)T_SEQ * B_BATCH * K_TAGS * sizeof(float));

  k_char_table<<<128, 256, 0, stream>>>(char_emb, char_w_ih, char_b, CT);
  k_cvt_f16<<<(256 * 64 + 255) / 256, 256, 0, stream>>>(char_w_hh, whhC16,
                                                        256 * 64);
  k_pack_whh_frag<<<(1024 * 256 + 255) / 256, 256, 0, stream>>>(word_w_hh,
                                                                whhT16);
  k_pack_wih<<<(1024 * KPAD + 255) / 256, 256, 0, stream>>>(word_w_ih, wihP16);
  k_pack_x<<<(NWORDS * KPAD + 255) / 256, 256, 0, stream>>>(word_inputs, word_emb,
                                                            Xpad);
  k_char_lstm<<<128, 256, 0, stream>>>(char_inputs, CT, whhC16, Xpad);
  k_word_gates<<<512, 256, 0, stream>>>(Xpad, wihP16, word_b, WG);
  k_word_lstm<<<1, 1024, 0, stream>>>(WG, whhT16, proj_w, proj_b, logits);
  k_crf<<<1, 64, 0, stream>>>(logits, tag_inputs, trans, start_t, end_t,
                              (float*)d_out);
}
